// FridaDistillationModel_3599182594770
// MI455X (gfx1250) — matrix operand order repaired and checked by reading, not compiled, not run on hardware
//
#include <hip/hip_runtime.h>
#include <hip/hip_bf16.h>

typedef __bf16 bf16;
typedef __attribute__((ext_vector_type(16))) __bf16 v16bf;
typedef __attribute__((ext_vector_type(8)))  float  v8f;

// Model dims
#define DV   50000
#define DT   1536
#define DH   368
#define DHP  384      // H padded to mult of 32 (K dim for GEMMs over H)
#define DL   3
#define DFF  1472
#define DNH  8
#define DHD  46
#define DHDP 64       // head dim padded to mult of 32
#define DB   4
#define DS   2048
#define DM   (DB*DS)  // 8192 tokens
#define EPS  1e-5f

union ABu { v16bf v; uint4 q[2]; };

__device__ __forceinline__ v16bf load_bf16x16(const bf16* p) {
    ABu u;
    const uint4* q = (const uint4*)p;
    u.q[0] = q[0];
    u.q[1] = q[1];
    return u.v;
}

// A fragment (16x32 bf16) from row-major A with leading dim lda.
// Lane l: row m0 + (l&15); slot s -> k = k0 + 16h + s
__device__ __forceinline__ v16bf load_afrag(const bf16* A, int lda, int m0, int k0, int lane) {
    int h = lane >> 4, m = lane & 15;
    const bf16* p = A + (size_t)(m0 + m) * lda + k0 + 16 * h;
    ABu u;
    u.q[0] = *(const uint4*)(p);
    u.q[1] = *(const uint4*)(p + 8);
    return u.v;
}

__device__ __forceinline__ v8f wmma_bf16(v16bf a, v16bf b, v8f c) {
    return __builtin_amdgcn_wmma_f32_16x16x32_bf16(false, a, false, b, (short)0, c, false, false);
}

// ---------------- 16-lane xor reductions (row-local) via DPP row_xmask ----------------
#if defined(__has_builtin) && __has_builtin(__builtin_amdgcn_update_dpp)
template <int CTRL>
__device__ __forceinline__ float dpp_maxstep(float v) {
    return fmaxf(v, __int_as_float(__builtin_amdgcn_update_dpp(
        0, __float_as_int(v), CTRL, 0xf, 0xf, true)));
}
template <int CTRL>
__device__ __forceinline__ float dpp_addstep(float v) {
    return v + __int_as_float(__builtin_amdgcn_update_dpp(
        0, __float_as_int(v), CTRL, 0xf, 0xf, true));
}
__device__ __forceinline__ float rowmax16(float v) {
    v = dpp_maxstep<0x161>(v);   // row_xmask:1
    v = dpp_maxstep<0x162>(v);   // row_xmask:2
    v = dpp_maxstep<0x164>(v);   // row_xmask:4
    v = dpp_maxstep<0x168>(v);   // row_xmask:8
    return v;
}
__device__ __forceinline__ float rowsum16(float v) {
    v = dpp_addstep<0x161>(v);
    v = dpp_addstep<0x162>(v);
    v = dpp_addstep<0x164>(v);
    v = dpp_addstep<0x168>(v);
    return v;
}
#else
__device__ __forceinline__ float rowmax16(float v) {
    for (int off = 1; off < 16; off <<= 1) v = fmaxf(v, __shfl_xor(v, off, 32));
    return v;
}
__device__ __forceinline__ float rowsum16(float v) {
    for (int off = 1; off < 16; off <<= 1) v += __shfl_xor(v, off, 32);
    return v;
}
#endif

// ---------------- weight packing: fp32 W[K,N] -> bf16 B-fragment-major ----------------
// block (nt,kt) holds 512 bf16: index = ((nt*Kt + kt)*32 + lane)*16 + s
// value = W[kt*32 + 16h + s, nt*16 + (lane&15)]  (zero-padded out of range)
__global__ __launch_bounds__(256) void pack_w_kernel(const float* __restrict__ W, bf16* __restrict__ out,
                                                     int K, int N, int Kt, int Ntiles) {
    int total = Ntiles * Kt * 512;
    for (int idx = blockIdx.x * blockDim.x + threadIdx.x; idx < total; idx += gridDim.x * blockDim.x) {
        int s  = idx & 15;
        int l  = (idx >> 4) & 31;
        int bl = idx >> 9;
        int kt = bl % Kt;
        int nt = bl / Kt;
        int h  = l >> 4;
        int n  = nt * 16 + (l & 15);
        int k  = kt * 32 + 16 * h + s;
        float v = (k < K && n < N) ? W[(size_t)k * N + n] : 0.0f;
        out[idx] = (bf16)v;
    }
}

// ---------------- embedding gather -> bf16 A matrix [M, T] ----------------
__global__ __launch_bounds__(256) void embed_kernel(const int* __restrict__ ids,
                                                    const float* __restrict__ emb,
                                                    bf16* __restrict__ out) {
    const int total = DM * DT;
    for (int idx = blockIdx.x * blockDim.x + threadIdx.x; idx < total; idx += gridDim.x * blockDim.x) {
        int row = idx / DT;
        int c   = idx - row * DT;
        int id  = ids[row];
        out[idx] = (bf16)emb[(size_t)id * DT + c];
    }
}

// ---------------- LayerNorm: fp32 x[M,368] -> bf16 [M,384] (padded) or fp32 out[M,368] ----------------
__global__ __launch_bounds__(256) void ln_kernel(const float* __restrict__ x,
                                                 const float* __restrict__ w,
                                                 const float* __restrict__ b,
                                                 bf16* __restrict__ outB,
                                                 float* __restrict__ outF) {
    int row = blockIdx.x * (blockDim.x >> 5) + (threadIdx.x >> 5);
    if (row >= DM) return;
    int lane = threadIdx.x & 31;
    float v[12];
    float sum = 0.f;
#pragma unroll
    for (int i = 0; i < 12; i++) {
        int c = lane + 32 * i;
        v[i] = (c < DH) ? x[(size_t)row * DH + c] : 0.f;
        sum += v[i];
    }
#pragma unroll
    for (int off = 1; off < 32; off <<= 1) sum += __shfl_xor(sum, off, 32);
    float mean = sum * (1.0f / DH);
    float s2 = 0.f;
#pragma unroll
    for (int i = 0; i < 12; i++) {
        int c = lane + 32 * i;
        if (c < DH) { float d = v[i] - mean; s2 += d * d; }
    }
#pragma unroll
    for (int off = 1; off < 32; off <<= 1) s2 += __shfl_xor(s2, off, 32);
    float rs = rsqrtf(s2 * (1.0f / DH) + EPS);
#pragma unroll
    for (int i = 0; i < 12; i++) {
        int c = lane + 32 * i;
        if (c < DH) {
            float val = (v[i] - mean) * rs * w[c] + b[c];
            if (outB) outB[(size_t)row * DHP + c] = (bf16)val;
            else      outF[(size_t)row * DH  + c] = val;
        }
    }
    if (outB && lane < 16) outB[(size_t)row * DHP + DH + lane] = (bf16)0.0f;
}

// ---------------- generic WMMA GEMM: 64x64 tile per wave ----------------
// MODE 0: fp32 store            (proj)
// MODE 1: fp32 add-in-place     (out-proj / fc2 residual)
// MODE 2: GELU -> bf16 [M,N]    (fc1)
// MODE 3: bf16 -> Q/K layout [bh,S,HDP]
// MODE 4: bf16 -> Vt layout [bh,HDP,S]
template <int MODE>
__global__ __launch_bounds__(256) void gemm_kernel(const bf16* __restrict__ A, int lda, int Kt,
                                                   const bf16* __restrict__ Wp,
                                                   const float* __restrict__ bias,
                                                   int Ntiles, int N,
                                                   float* __restrict__ outF,
                                                   bf16* __restrict__ outB) {
    int lane = threadIdx.x & 31;
    int wv   = threadIdx.x >> 5;
    int task = blockIdx.x * (blockDim.x >> 5) + wv;
    const int Mb = DM / 64;                      // 128 row-blocks of 64
    int nstrips = (Ntiles + 3) >> 2;
    if (task >= Mb * nstrips) return;            // wave-uniform
    int mb = task % Mb;
    int ns = task / Mb;
    int m0 = mb * 64;

    v8f zero = {0.f, 0.f, 0.f, 0.f, 0.f, 0.f, 0.f, 0.f};
    v8f acc[4][4];
#pragma unroll
    for (int mi = 0; mi < 4; mi++)
#pragma unroll
        for (int i = 0; i < 4; i++) acc[mi][i] = zero;

    for (int kt = 0; kt < Kt; kt++) {
        v16bf bfr[4];
#pragma unroll
        for (int i = 0; i < 4; i++) {
            int nt = ns * 4 + i;
            if (nt < Ntiles)                     // wave-uniform
                bfr[i] = load_bf16x16(Wp + (((size_t)nt * Kt + kt) << 9) + lane * 16);
        }
#pragma unroll
        for (int mi = 0; mi < 4; mi++) {
            v16bf a = load_afrag(A, lda, m0 + 16 * mi, kt * 32, lane);
#pragma unroll
            for (int i = 0; i < 4; i++) {
                int nt = ns * 4 + i;
                if (nt < Ntiles)
                    acc[mi][i] = wmma_bf16(a, bfr[i], acc[mi][i]);
            }
        }
    }

    int h = lane >> 4, ln = lane & 15;
#pragma unroll
    for (int i = 0; i < 4; i++) {
        int nt = ns * 4 + i;
        if (nt >= Ntiles) continue;
        int n = nt * 16 + ln;
        float bs = bias[n];
#pragma unroll
        for (int mi = 0; mi < 4; mi++) {
#pragma unroll
            for (int r = 0; r < 8; r++) {
                int m = m0 + 16 * mi + r + 8 * h;
                float c = acc[mi][i][r] + bs;
                if (MODE == 0) {
                    outF[(size_t)m * N + n] = c;
                } else if (MODE == 1) {
                    outF[(size_t)m * N + n] += c;
                } else if (MODE == 2) {
                    float g = 0.5f * c * (1.0f + erff(c * 0.70710678118f));
                    outB[(size_t)m * N + n] = (bf16)g;
                } else if (MODE == 3) {
                    int head = n / DHD, hd = n - head * DHD;
                    int bb = m >> 11, s = m & (DS - 1);
                    outB[(((size_t)(bb * DNH + head) * DS + s) * DHDP) + hd] = (bf16)c;
                } else if (MODE == 4) {
                    int head = n / DHD, hd = n - head * DHD;
                    int bb = m >> 11, s = m & (DS - 1);
                    outB[(((size_t)(bb * DNH + head) * DHDP + hd) * DS) + s] = (bf16)c;
                }
            }
        }
    }
}

// ---------------- flash attention: one wave per (bh, 32-query tile) ----------------
__global__ __launch_bounds__(256) void attn_kernel(const bf16* __restrict__ Q,
                                                   const bf16* __restrict__ K,
                                                   const bf16* __restrict__ Vt,
                                                   const float* __restrict__ mask,
                                                   bf16* __restrict__ ctxA) {
    __shared__ bf16 plds[8][2][512];
    int lane = threadIdx.x & 31;
    int wv   = threadIdx.x >> 5;
    int id   = blockIdx.x * 8 + wv;             // 2048 tasks exactly (256 blocks)
    int bh   = id >> 6;
    int qt   = id & 63;
    int b    = bh >> 3;
    int head = bh & 7;
    int h = lane >> 4, ln = lane & 15;
    int q0 = qt * 32;

    const bf16* Qb = Q  + (size_t)bh * DS * DHDP;
    const bf16* Kb = K  + (size_t)bh * DS * DHDP;
    const bf16* Vb = Vt + (size_t)bh * DHDP * DS;

    v16bf qa[2][2];
#pragma unroll
    for (int mi = 0; mi < 2; mi++) {
        qa[mi][0] = load_afrag(Qb, DHDP, q0 + 16 * mi, 0, lane);
        qa[mi][1] = load_afrag(Qb, DHDP, q0 + 16 * mi, 32, lane);
    }

    v8f zero = {0.f, 0.f, 0.f, 0.f, 0.f, 0.f, 0.f, 0.f};
    v8f acc[2][4];
    float mrun[2][8], lrun[2][8];
#pragma unroll
    for (int mi = 0; mi < 2; mi++) {
#pragma unroll
        for (int i = 0; i < 4; i++) acc[mi][i] = zero;
#pragma unroll
        for (int r = 0; r < 8; r++) { mrun[mi][r] = -3.0e38f; lrun[mi][r] = 0.f; }
    }

    const float scale = 0.147441956155f; // 1/sqrt(46)

    for (int kt2 = 0; kt2 < DS; kt2 += 32) {
        // K fragments for this 32-key tile, shared by both query tiles
        v16bf kb[2][2];
#pragma unroll
        for (int j = 0; j < 2; j++) {
            int key0 = kt2 + 16 * j;
            kb[j][0] = load_bf16x16(Kb + (size_t)(key0 + ln) * DHDP + 16 * h);
            kb[j][1] = load_bf16x16(Kb + (size_t)(key0 + ln) * DHDP + 32 + 16 * h);
        }
        float am0 = (1.0f - mask[b * DS + kt2 + ln]) * -10000.0f;
        float am1 = (1.0f - mask[b * DS + kt2 + 16 + ln]) * -10000.0f;

#pragma unroll
        for (int mi = 0; mi < 2; mi++) {
            v8f sc0 = zero, sc1 = zero;
            sc0 = wmma_bf16(qa[mi][0], kb[0][0], sc0);
            sc0 = wmma_bf16(qa[mi][1], kb[0][1], sc0);
            sc1 = wmma_bf16(qa[mi][0], kb[1][0], sc1);
            sc1 = wmma_bf16(qa[mi][1], kb[1][1], sc1);

            float alpha[8];
#pragma unroll
            for (int r = 0; r < 8; r++) {
                float s0 = sc0[r] * scale + am0;
                float s1 = sc1[r] * scale + am1;
                float mx = rowmax16(fmaxf(s0, s1));
                float nm = fmaxf(mrun[mi][r], mx);
                float al = __expf(mrun[mi][r] - nm);
                float p0 = __expf(s0 - nm);
                float p1 = __expf(s1 - nm);
                float rs = rowsum16(p0 + p1);
                lrun[mi][r] = lrun[mi][r] * al + rs;
                mrun[mi][r] = nm;
                alpha[r] = al;
                int m = r + 8 * h;
                plds[wv][mi][m * 32 + ln]      = (bf16)p0;
                plds[wv][mi][m * 32 + 16 + ln] = (bf16)p1;
            }
#pragma unroll
            for (int f = 0; f < 4; f++)
#pragma unroll
                for (int r = 0; r < 8; r++) acc[mi][f][r] *= alpha[r];
        }

        asm volatile("s_wait_dscnt 0" ::: "memory");

        // P tiles (16x32 bf16) as A fragments from LDS
        v16bf pA[2];
#pragma unroll
        for (int mi = 0; mi < 2; mi++) {
            ABu u;
            const bf16* pb = &plds[wv][mi][ln * 32 + 16 * h];
            u.q[0] = *(const uint4*)(pb);
            u.q[1] = *(const uint4*)(pb + 8);
            pA[mi] = u.v;
        }

#pragma unroll
        for (int f = 0; f < 4; f++) {
            v16bf vbf = load_bf16x16(Vb + (size_t)(16 * f + ln) * DS + kt2 + 16 * h);
            acc[0][f] = wmma_bf16(pA[0], vbf, acc[0][f]);
            acc[1][f] = wmma_bf16(pA[1], vbf, acc[1][f]);
        }
    }

    // normalize + scatter ctx (bf16, [token, 384] padded layout, col = head*46+hd)
#pragma unroll
    for (int mi = 0; mi < 2; mi++) {
#pragma unroll
        for (int f = 0; f < 4; f++) {
            int hd = 16 * f + ln;
#pragma unroll
            for (int r = 0; r < 8; r++) {
                if (hd < DHD) {
                    float val = acc[mi][f][r] / lrun[mi][r];
                    int tok = b * DS + q0 + 16 * mi + r + 8 * h;
                    ctxA[(size_t)tok * DHP + head * DHD + hd] = (bf16)val;
                }
            }
        }
    }
}

// ---------------- host ----------------
static inline void* bump(char* base, size_t& off, size_t bytes) {
    void* p = base + off;
    off += (bytes + 255) & ~(size_t)255;
    return p;
}

extern "C" void kernel_launch(void* const* d_in, const int* in_sizes, int n_in,
                              void* d_out, int out_size, void* d_ws, size_t ws_size,
                              hipStream_t stream) {
    const int*   ids    = (const int*)d_in[0];
    const float* mask   = (const float*)d_in[1];
    const float* emb    = (const float*)d_in[2];
    const float* proj_w = (const float*)d_in[3];
    const float* proj_b = (const float*)d_in[4];
    const float* ln1_w  = (const float*)d_in[5];
    const float* ln1_b  = (const float*)d_in[6];
    const float* qw     = (const float*)d_in[7];
    const float* qb     = (const float*)d_in[8];
    const float* kw     = (const float*)d_in[9];
    const float* kb     = (const float*)d_in[10];
    const float* vw     = (const float*)d_in[11];
    const float* vb     = (const float*)d_in[12];
    const float* ow     = (const float*)d_in[13];
    const float* ob     = (const float*)d_in[14];
    const float* ln2_w  = (const float*)d_in[15];
    const float* ln2_b  = (const float*)d_in[16];
    const float* fc1_w  = (const float*)d_in[17];
    const float* fc1_b  = (const float*)d_in[18];
    const float* fc2_w  = (const float*)d_in[19];
    const float* fc2_b  = (const float*)d_in[20];
    const float* lnf_w  = (const float*)d_in[21];
    const float* lnf_b  = (const float*)d_in[22];
    float* out = (float*)d_out;

    char* ws = (char*)d_ws;
    size_t off = 0;
    bf16*  embA = (bf16*)bump(ws, off, (size_t)DM * DT * 2);
    float* x    = (float*)bump(ws, off, (size_t)DM * DH * 4);
    bf16*  hA   = (bf16*)bump(ws, off, (size_t)DM * DHP * 2);
    bf16*  ctxA = (bf16*)bump(ws, off, (size_t)DM * DHP * 2);
    bf16*  ffA  = (bf16*)bump(ws, off, (size_t)DM * DFF * 2);
    size_t qk_bytes = (size_t)DB * DNH * DS * DHDP * 2;
    bf16*  Qb   = (bf16*)bump(ws, off, qk_bytes);
    bf16*  Kb   = (bf16*)bump(ws, off, qk_bytes);
    bf16*  Vtb  = (bf16*)bump(ws, off, qk_bytes);

    // packed weights
    const int KtT = DT / 32, KtH = DHP / 32, KtF = DFF / 32;
    const int NtH = DH / 16, NtF = DFF / 16;
    bf16* pk_proj = (bf16*)bump(ws, off, (size_t)NtH * KtT * 512 * 2);
    bf16 *pk_q[DL], *pk_k[DL], *pk_v[DL], *pk_o[DL], *pk_f1[DL], *pk_f2[DL];
    for (int i = 0; i < DL; i++) {
        pk_q[i]  = (bf16*)bump(ws, off, (size_t)NtH * KtH * 512 * 2);
        pk_k[i]  = (bf16*)bump(ws, off, (size_t)NtH * KtH * 512 * 2);
        pk_v[i]  = (bf16*)bump(ws, off, (size_t)NtH * KtH * 512 * 2);
        pk_o[i]  = (bf16*)bump(ws, off, (size_t)NtH * KtH * 512 * 2);
        pk_f1[i] = (bf16*)bump(ws, off, (size_t)NtF * KtH * 512 * 2);
        pk_f2[i] = (bf16*)bump(ws, off, (size_t)NtH * KtF * 512 * 2);
    }

    // pack all weights
    pack_w_kernel<<<2048, 256, 0, stream>>>(proj_w, pk_proj, DT, DH, KtT, NtH);
    for (int i = 0; i < DL; i++) {
        pack_w_kernel<<<1024, 256, 0, stream>>>(qw + (size_t)i * DH * DH, pk_q[i], DH, DH, KtH, NtH);
        pack_w_kernel<<<1024, 256, 0, stream>>>(kw + (size_t)i * DH * DH, pk_k[i], DH, DH, KtH, NtH);
        pack_w_kernel<<<1024, 256, 0, stream>>>(vw + (size_t)i * DH * DH, pk_v[i], DH, DH, KtH, NtH);
        pack_w_kernel<<<1024, 256, 0, stream>>>(ow + (size_t)i * DH * DH, pk_o[i], DH, DH, KtH, NtH);
        pack_w_kernel<<<2048, 256, 0, stream>>>(fc1_w + (size_t)i * DH * DFF, pk_f1[i], DH, DFF, KtH, NtF);
        pack_w_kernel<<<2048, 256, 0, stream>>>(fc2_w + (size_t)i * DFF * DH, pk_f2[i], DFF, DH, KtF, NtH);
    }

    // embedding gather
    embed_kernel<<<8192, 256, 0, stream>>>(ids, emb, embA);

    // zero the padded bf16 buffers (pads must be exact zeros)
    hipMemsetAsync(Qb, 0, qk_bytes, stream);
    hipMemsetAsync(Kb, 0, qk_bytes, stream);
    hipMemsetAsync(Vtb, 0, qk_bytes, stream);
    hipMemsetAsync(ctxA, 0, (size_t)DM * DHP * 2, stream);

    const int Mb = DM / 64;
    auto gblocks = [&](int Ntiles) { return (Mb * ((Ntiles + 3) / 4) + 7) / 8; };

    // projection: x = embA @ proj_w + b
    gemm_kernel<0><<<gblocks(NtH), 256, 0, stream>>>(embA, DT, KtT, pk_proj, proj_b, NtH, DH, x, nullptr);

    for (int i = 0; i < DL; i++) {
        ln_kernel<<<1024, 256, 0, stream>>>(x, ln1_w + i * DH, ln1_b + i * DH, hA, nullptr);
        gemm_kernel<3><<<gblocks(NtH), 256, 0, stream>>>(hA, DHP, KtH, pk_q[i], qb + i * DH, NtH, DH, nullptr, Qb);
        gemm_kernel<3><<<gblocks(NtH), 256, 0, stream>>>(hA, DHP, KtH, pk_k[i], kb + i * DH, NtH, DH, nullptr, Kb);
        gemm_kernel<4><<<gblocks(NtH), 256, 0, stream>>>(hA, DHP, KtH, pk_v[i], vb + i * DH, NtH, DH, nullptr, Vtb);
        attn_kernel<<<256, 256, 0, stream>>>(Qb, Kb, Vtb, mask, ctxA);
        gemm_kernel<1><<<gblocks(NtH), 256, 0, stream>>>(ctxA, DHP, KtH, pk_o[i], ob + i * DH, NtH, DH, x, nullptr);
        ln_kernel<<<1024, 256, 0, stream>>>(x, ln2_w + i * DH, ln2_b + i * DH, hA, nullptr);
        gemm_kernel<2><<<gblocks(NtF), 256, 0, stream>>>(hA, DHP, KtH, pk_f1[i], fc1_b + i * DFF, NtF, DFF, nullptr, ffA);
        gemm_kernel<1><<<gblocks(NtH), 256, 0, stream>>>(ffA, DFF, KtF, pk_f2[i], fc2_b + i * DH, NtH, DH, x, nullptr);
    }

    ln_kernel<<<1024, 256, 0, stream>>>(x, lnf_w, lnf_b, nullptr, out);
}
